// subLSTMCell_52587579572453
// MI455X (gfx1250) — matrix operand order repaired and checked by reading, not compiled, not run on hardware
//
#include <hip/hip_runtime.h>

// ---------------------------------------------------------------------------
// subLSTM cell for MI455X (gfx1250), bf16 WMMA.
// B=8192, D_IN=1024, UNITS=1024, D_OUT=1024.
//
// Fast path (needs 18.9 MB workspace):
//   P1: transpose+convert [Wx;Wh] -> bf16 Wt[4096][2048] (K-contiguous),
//       Wc -> bf16 Wct[1024][1024].
//   K1: gates GEMM (K=2048, fused x+h) with operands loaded straight from
//       global in WMMA register layout (no LDS, no barriers), fused
//       sigmoid/cell epilogue in registers -> c, h_new.
//   K2: outputs = h_new @ Wc + bc, same scheme.
// Fallback path (small ws): LDS-staged kernels.
// ---------------------------------------------------------------------------

typedef __attribute__((ext_vector_type(16))) __bf16        v16bf;
typedef __attribute__((ext_vector_type(8)))  float         v8f;
typedef __attribute__((ext_vector_type(4)))  float         v4f;
typedef __attribute__((ext_vector_type(4)))  unsigned int  v4u;

#define BATCH 8192
#define DIN   1024
#define UNITS 1024
#define NGATE 4
#define KCAT  2048   // D_IN + UNITS (fused gate GEMM K)
#define ASTR  40     // fallback LDS row stride (bf16 elems, 80 B)

// float pair -> packed bf16x2, round-to-nearest-even.
// Rounded high halves merged with a single v_perm_b32.
__device__ __forceinline__ unsigned int pack_bf16x2(float a, float b) {
    unsigned int ua = __float_as_uint(a);
    unsigned int ub = __float_as_uint(b);
    ua += 0x7FFFu + ((ua >> 16) & 1u);
    ub += 0x7FFFu + ((ub >> 16) & 1u);
    // dst = { ub[31:24], ub[23:16], ua[31:24], ua[23:16] }
    return __builtin_amdgcn_perm(ub, ua, 0x07060302u);
}

__device__ __forceinline__ unsigned short f2bf(float a) {
    unsigned int ua = __float_as_uint(a);
    ua += 0x7FFFu + ((ua >> 16) & 1u);
    return (unsigned short)(ua >> 16);
}

union Frag {
    struct { v4u lo, hi; } u;
    v16bf bf;
};

__device__ __forceinline__ float sigmoidf(float x) {
    return 1.0f / (1.0f + __expf(-x));
}

// A fragment straight from global fp32 (row-major, row stride = ld floats).
// p must already include: row*ld + kb + (lane>=16 ? 8 : 0).
// lanes 0-15: V0-3=K0-7, V4-7=K16-23 ; lanes 16-31: V0-3=K8-15, V4-7=K24-31
__device__ __forceinline__ v16bf load_a_global(const float* __restrict__ p) {
    v4f f0 = *(const v4f*)(p);
    v4f f1 = *(const v4f*)(p + 4);
    v4f f2 = *(const v4f*)(p + 16);
    v4f f3 = *(const v4f*)(p + 20);
    Frag fr;
    fr.u.lo = (v4u){pack_bf16x2(f0.x, f0.y), pack_bf16x2(f0.z, f0.w),
                    pack_bf16x2(f1.x, f1.y), pack_bf16x2(f1.z, f1.w)};
    fr.u.hi = (v4u){pack_bf16x2(f2.x, f2.y), pack_bf16x2(f2.z, f2.w),
                    pack_bf16x2(f3.x, f3.y), pack_bf16x2(f3.z, f3.w)};
    return fr.bf;
}

// B fragment from pre-transposed bf16 weights (column-major over K).
// p must already include: col*Kstride + kb + (lane>=16 ? 8 : 0).
// lanes 0-15: V0-3=K0-7, V4-7=K16-23 ; lanes 16-31: V0-3=K8-15, V4-7=K24-31
__device__ __forceinline__ v16bf load_b_global(const unsigned short* __restrict__ p) {
    Frag fr;
    fr.u.lo = *(const v4u*)(p);
    fr.u.hi = *(const v4u*)(p + 16);
    return fr.bf;
}

// ---------------------------------------------------------------------------
// P1: transpose + convert weights: in[k][n] fp32 -> out[n*outStride + kBase + k]
// One thread produces 8 consecutive K of one column (one b128 store).
// ---------------------------------------------------------------------------
__global__ __launch_bounds__(256) void wconv_kernel(
    const float* __restrict__ in, unsigned short* __restrict__ out,
    int Ksrc, int N, int outStride, int kBase)
{
    const int t = blockIdx.x * blockDim.x + threadIdx.x;
    const int total = (Ksrc >> 3) * N;
    if (t >= total) return;
    const int n  = t % N;
    const int k0 = (t / N) * 8;
    float f[8];
    #pragma unroll
    for (int i = 0; i < 8; ++i) f[i] = in[(size_t)(k0 + i) * N + n];
    v4u pk = {pack_bf16x2(f[0], f[1]), pack_bf16x2(f[2], f[3]),
              pack_bf16x2(f[4], f[5]), pack_bf16x2(f[6], f[7])};
    *(v4u*)(out + (size_t)n * outStride + kBase + k0) = pk;
}

// ---------------------------------------------------------------------------
// K1: fused gate GEMM + cell update. Block: 128 rows x 32 unit-cols x 4 gates.
// 8 waves: wm=wave&3 -> 32-row group (2 subtiles of 16), wn=wave>>2 -> 16-col half.
// Grid (64, 32). No LDS, no barriers.
// ---------------------------------------------------------------------------
__global__ __launch_bounds__(256) void sublstm_gates_wmma(
    const float* __restrict__ inputs, const float* __restrict__ h_tm1,
    const float* __restrict__ c_tm1,
    const unsigned short* __restrict__ Wt,   // [4*UNITS][KCAT] bf16
    const float* __restrict__ bx, const float* __restrict__ bh,
    float* __restrict__ c_out, float* __restrict__ h_out)
{
    const int tid  = threadIdx.x;
    const int lane = tid & 31;
    const int wave = tid >> 5;
    const int wm   = wave & 3;
    const int wn   = wave >> 2;
    const int m0   = blockIdx.x * 128;
    const int n0   = blockIdx.y * 32;

    const int half = lane >> 4;
    const int l15  = lane & 15;

    const int rowA0 = m0 + wm * 32 + l15;        // subtile 0
    const int rowA1 = rowA0 + 16;                // subtile 1
    const int col   = n0 + wn * 16 + l15;

    const unsigned short* bptr[NGATE];
    #pragma unroll
    for (int g = 0; g < NGATE; ++g)
        bptr[g] = Wt + (size_t)(g * UNITS + col) * KCAT + half * 8;

    v8f acc[2][NGATE] = {};

    for (int phase = 0; phase < 2; ++phase) {
        const float* __restrict__ Asrc = phase ? h_tm1 : inputs;
        const int koff = phase * DIN;
        const float* a0p = Asrc + (size_t)rowA0 * DIN + half * 8;
        const float* a1p = Asrc + (size_t)rowA1 * DIN + half * 8;

        for (int kb = 0; kb < DIN; kb += 32) {
            // issue all operand loads first (distinct regs -> one deep clause,
            // 8 B-loads + 8 A-loads in flight, overlapped with prior WMMAs)
            v16bf b[NGATE];
            #pragma unroll
            for (int g = 0; g < NGATE; ++g)
                b[g] = load_b_global(bptr[g] + koff + kb);
            const v16bf a0 = load_a_global(a0p + kb);
            const v16bf a1 = load_a_global(a1p + kb);

            #pragma unroll
            for (int g = 0; g < NGATE; ++g) {
                acc[0][g] = __builtin_amdgcn_wmma_f32_16x16x32_bf16(
                    false, a0, false, b[g], (short)0, acc[0][g], false, false);
                acc[1][g] = __builtin_amdgcn_wmma_f32_16x16x32_bf16(
                    false, a1, false, b[g], (short)0, acc[1][g], false, false);
            }
        }
    }

    // fused epilogue: gates of one (row,col) share a lane/VGPR slot.
    float bias[NGATE];
    #pragma unroll
    for (int g = 0; g < NGATE; ++g)
        bias[g] = bx[g * UNITS + col] + bh[g * UNITS + col];

    #pragma unroll
    for (int sub = 0; sub < 2; ++sub) {
        const int rbase = m0 + wm * 32 + sub * 16 + half * 8;
        #pragma unroll
        for (int v = 0; v < 8; ++v) {
            const size_t off = (size_t)(rbase + v) * UNITS + col;
            const float ig = sigmoidf(acc[sub][0][v] + bias[0]);
            const float fg = sigmoidf(acc[sub][1][v] + bias[1]);
            const float cg = sigmoidf(acc[sub][2][v] + bias[2]);
            const float og = sigmoidf(acc[sub][3][v] + bias[3]);
            const float c  = fg * c_tm1[off] + cg - ig;
            const float hn = sigmoidf(c) - og;
            c_out[off] = c;
            h_out[off] = hn;
        }
    }
}

// ---------------------------------------------------------------------------
// K2: outputs = h_new @ Wc + bc. Block: 128 rows x 64 cols. Grid (64, 16).
// ---------------------------------------------------------------------------
__global__ __launch_bounds__(256) void sublstm_out_wmma(
    const float* __restrict__ h_new,
    const unsigned short* __restrict__ Wct,  // [UNITS][UNITS] bf16, K-contig
    const float* __restrict__ bc, float* __restrict__ out)
{
    const int tid  = threadIdx.x;
    const int lane = tid & 31;
    const int wave = tid >> 5;
    const int wm   = wave & 3;
    const int wn   = wave >> 2;
    const int m0   = blockIdx.x * 128;
    const int n0   = blockIdx.y * 64;

    const int half = lane >> 4;
    const int l15  = lane & 15;

    const int rowA0 = m0 + wm * 32 + l15;
    const int rowA1 = rowA0 + 16;

    const unsigned short* bptr[2];
    #pragma unroll
    for (int t = 0; t < 2; ++t)
        bptr[t] = Wct + (size_t)(n0 + wn * 32 + t * 16 + l15) * UNITS + half * 8;

    const float* a0p = h_new + (size_t)rowA0 * UNITS + half * 8;
    const float* a1p = h_new + (size_t)rowA1 * UNITS + half * 8;

    v8f acc[2][2] = {};

    for (int kb = 0; kb < UNITS; kb += 32) {
        v16bf b[2];
        #pragma unroll
        for (int t = 0; t < 2; ++t)
            b[t] = load_b_global(bptr[t] + kb);
        const v16bf a0 = load_a_global(a0p + kb);
        const v16bf a1 = load_a_global(a1p + kb);

        #pragma unroll
        for (int t = 0; t < 2; ++t) {
            acc[0][t] = __builtin_amdgcn_wmma_f32_16x16x32_bf16(
                false, a0, false, b[t], (short)0, acc[0][t], false, false);
            acc[1][t] = __builtin_amdgcn_wmma_f32_16x16x32_bf16(
                false, a1, false, b[t], (short)0, acc[1][t], false, false);
        }
    }

    #pragma unroll
    for (int sub = 0; sub < 2; ++sub) {
        const int rbase = m0 + wm * 32 + sub * 16 + half * 8;
        #pragma unroll
        for (int t = 0; t < 2; ++t) {
            const int col = n0 + wn * 32 + t * 16 + l15;
            const float b = bc[col];
            #pragma unroll
            for (int v = 0; v < 8; ++v)
                out[(size_t)(rbase + v) * UNITS + col] = acc[sub][t][v] + b;
        }
    }
}

// ===========================================================================
// Fallback path (workspace too small): LDS-staged kernels.
// ===========================================================================
__device__ __forceinline__ v16bf load_a_frag(const unsigned short* Abuf, int wm, int lane) {
    const int r    = wm * 16 + (lane & 15);
    const int half = lane >> 4;
    const unsigned short* p = Abuf + r * ASTR;
    Frag f;
    f.u.lo = *(const v4u*)(p + half * 8);
    f.u.hi = *(const v4u*)(p + 16 + half * 8);
    return f.bf;
}

__device__ __forceinline__ v16bf load_b_frag(const unsigned short* colbase, int lane) {
    const int c    = lane & 15;
    const int half = lane >> 4;
    const unsigned short* p = colbase + c * ASTR;
    Frag f;
    f.u.lo = *(const v4u*)(p + half * 8);
    f.u.hi = *(const v4u*)(p + 16 + half * 8);
    return f.bf;
}

__global__ __launch_bounds__(256) void sublstm_gates_fb(
    const float* __restrict__ inputs, const float* __restrict__ h_tm1,
    const float* __restrict__ c_tm1,
    const float* __restrict__ Wx, const float* __restrict__ bx,
    const float* __restrict__ Wh, const float* __restrict__ bh,
    float* __restrict__ c_out, float* __restrict__ h_out)
{
    __shared__ __align__(16) unsigned short Abuf[64 * ASTR];
    __shared__ __align__(16) unsigned short Bbuf[NGATE * 32 * ASTR];

    const int tid  = threadIdx.x;
    const int lane = tid & 31;
    const int wave = tid >> 5;
    const int wm   = wave & 3;
    const int wn   = wave >> 2;
    const int m0   = blockIdx.x * 64;
    const int n0   = blockIdx.y * 32;
    const int a_row = tid >> 2;
    const int a_col = (tid & 3) * 8;

    v8f acc[NGATE] = {};

    for (int phase = 0; phase < 2; ++phase) {
        const float* __restrict__ Asrc = phase ? h_tm1 : inputs;
        const float* __restrict__ Wsrc = phase ? Wh : Wx;
        for (int kb = 0; kb < DIN; kb += 32) {
            {
                const float* p = Asrc + (size_t)(m0 + a_row) * DIN + kb + a_col;
                v4f f0 = *(const v4f*)p;
                v4f f1 = *(const v4f*)(p + 4);
                v4u pk = {pack_bf16x2(f0.x, f0.y), pack_bf16x2(f0.z, f0.w),
                          pack_bf16x2(f1.x, f1.y), pack_bf16x2(f1.z, f1.w)};
                *(v4u*)(Abuf + a_row * ASTR + a_col) = pk;
            }
            #pragma unroll
            for (int i = 0; i < 4; ++i) {
                const int idx = tid + i * 256;
                const int k   = idx >> 5;
                const int nc4 = (idx & 31) * 4;
                const int g   = nc4 >> 5;
                const int nc  = nc4 & 31;
                const float* p = Wsrc + (size_t)(kb + k) * (NGATE * UNITS)
                                 + g * UNITS + n0 + nc;
                v4f f = *(const v4f*)p;
                unsigned short* q = Bbuf + (g * 32 + nc) * ASTR + k;
                q[0]        = f2bf(f.x);
                q[ASTR]     = f2bf(f.y);
                q[2 * ASTR] = f2bf(f.z);
                q[3 * ASTR] = f2bf(f.w);
            }
            __syncthreads();
            const v16bf a = load_a_frag(Abuf, wm, lane);
            #pragma unroll
            for (int g = 0; g < NGATE; ++g) {
                const v16bf b = load_b_frag(Bbuf + (g * 32 + wn * 16) * ASTR, lane);
                acc[g] = __builtin_amdgcn_wmma_f32_16x16x32_bf16(
                    false, a, false, b, (short)0, acc[g], false, false);
            }
            __syncthreads();
        }
    }

    const int col = n0 + wn * 16 + (lane & 15);
    float bias[NGATE];
    #pragma unroll
    for (int g = 0; g < NGATE; ++g)
        bias[g] = bx[g * UNITS + col] + bh[g * UNITS + col];

    const int rbase = m0 + wm * 16 + ((lane >> 4) << 3);
    #pragma unroll
    for (int v = 0; v < 8; ++v) {
        const size_t off = (size_t)(rbase + v) * UNITS + col;
        const float ig = sigmoidf(acc[0][v] + bias[0]);
        const float fg = sigmoidf(acc[1][v] + bias[1]);
        const float cg = sigmoidf(acc[2][v] + bias[2]);
        const float og = sigmoidf(acc[3][v] + bias[3]);
        const float c  = fg * c_tm1[off] + cg - ig;
        c_out[off] = c;
        h_out[off] = sigmoidf(c) - og;
    }
}

__global__ __launch_bounds__(256) void sublstm_out_fb(
    const float* __restrict__ h_new, const float* __restrict__ Wc,
    const float* __restrict__ bc, float* __restrict__ out)
{
    __shared__ __align__(16) unsigned short Abuf[64 * ASTR];
    __shared__ __align__(16) unsigned short Bbuf[64 * ASTR];

    const int tid  = threadIdx.x;
    const int lane = tid & 31;
    const int wave = tid >> 5;
    const int wm   = wave & 3;
    const int wn   = wave >> 2;
    const int m0   = blockIdx.x * 64;
    const int n0   = blockIdx.y * 64;
    const int a_row = tid >> 2;
    const int a_col = (tid & 3) * 8;

    v8f acc[2] = {};

    for (int kb = 0; kb < UNITS; kb += 32) {
        {
            const float* p = h_new + (size_t)(m0 + a_row) * UNITS + kb + a_col;
            v4f f0 = *(const v4f*)p;
            v4f f1 = *(const v4f*)(p + 4);
            v4u pk = {pack_bf16x2(f0.x, f0.y), pack_bf16x2(f0.z, f0.w),
                      pack_bf16x2(f1.x, f1.y), pack_bf16x2(f1.z, f1.w)};
            *(v4u*)(Abuf + a_row * ASTR + a_col) = pk;
        }
        #pragma unroll
        for (int i = 0; i < 2; ++i) {
            const int idx = tid + i * 256;
            const int k   = idx >> 4;
            const int nc  = (idx & 15) * 4;
            const float* p = Wc + (size_t)(kb + k) * UNITS + n0 + nc;
            v4f f = *(const v4f*)p;
            unsigned short* q = Bbuf + nc * ASTR + k;
            q[0]        = f2bf(f.x);
            q[ASTR]     = f2bf(f.y);
            q[2 * ASTR] = f2bf(f.z);
            q[3 * ASTR] = f2bf(f.w);
        }
        __syncthreads();
        const v16bf a = load_a_frag(Abuf, wm, lane);
        #pragma unroll
        for (int t = 0; t < 2; ++t) {
            const v16bf b = load_b_frag(Bbuf + (wn * 32 + t * 16) * ASTR, lane);
            acc[t] = __builtin_amdgcn_wmma_f32_16x16x32_bf16(
                false, a, false, b, (short)0, acc[t], false, false);
        }
        __syncthreads();
    }

    const int rbase = m0 + wm * 16 + ((lane >> 4) << 3);
    #pragma unroll
    for (int t = 0; t < 2; ++t) {
        const int col = n0 + wn * 32 + t * 16 + (lane & 15);
        const float b = bc[col];
        #pragma unroll
        for (int v = 0; v < 8; ++v)
            out[(size_t)(rbase + v) * UNITS + col] = acc[t][v] + b;
    }
}

// ---------------------------------------------------------------------------
extern "C" void kernel_launch(void* const* d_in, const int* in_sizes, int n_in,
                              void* d_out, int out_size, void* d_ws, size_t ws_size,
                              hipStream_t stream) {
    const float* inputs = (const float*)d_in[0];
    const float* h_tm1  = (const float*)d_in[1];
    const float* c_tm1  = (const float*)d_in[2];
    const float* Wx     = (const float*)d_in[3];
    const float* bx     = (const float*)d_in[4];
    const float* Wh     = (const float*)d_in[5];
    const float* bh     = (const float*)d_in[6];
    const float* Wc     = (const float*)d_in[7];
    const float* bc     = (const float*)d_in[8];

    float* out   = (float*)d_out;                        // outputs [B, D_OUT]
    float* h_new = out + (size_t)BATCH * UNITS;          // h_new   [B, UNITS]
    float* c_new = out + 2 * (size_t)BATCH * UNITS;      // c       [B, UNITS]

    const size_t wt_elems  = (size_t)NGATE * UNITS * KCAT;   // 8M bf16
    const size_t wct_elems = (size_t)UNITS * UNITS;          // 1M bf16
    const size_t ws_need   = (wt_elems + wct_elems) * sizeof(unsigned short);

    if (ws_size >= ws_need) {
        unsigned short* Wt  = (unsigned short*)d_ws;
        unsigned short* Wct = Wt + wt_elems;

        const int tx = (DIN / 8) * (NGATE * UNITS);          // 524288 threads
        const int tc = (UNITS / 8) * UNITS;                  // 131072 threads
        wconv_kernel<<<(tx + 255) / 256, 256, 0, stream>>>(Wx, Wt, DIN, NGATE * UNITS, KCAT, 0);
        wconv_kernel<<<(tx + 255) / 256, 256, 0, stream>>>(Wh, Wt, UNITS, NGATE * UNITS, KCAT, DIN);
        wconv_kernel<<<(tc + 255) / 256, 256, 0, stream>>>(Wc, Wct, UNITS, UNITS, UNITS, 0);

        dim3 g1(BATCH / 128, UNITS / 32);
        sublstm_gates_wmma<<<g1, 256, 0, stream>>>(
            inputs, h_tm1, c_tm1, Wt, bx, bh, c_new, h_new);

        dim3 g2(BATCH / 128, UNITS / 64);
        sublstm_out_wmma<<<g2, 256, 0, stream>>>(h_new, Wct, bc, out);
    } else {
        dim3 g1(BATCH / 64, UNITS / 32);
        sublstm_gates_fb<<<g1, 256, 0, stream>>>(
            inputs, h_tm1, c_tm1, Wx, bx, Wh, bh, c_new, h_new);
        dim3 g2(BATCH / 64, UNITS / 64);
        sublstm_out_fb<<<g2, 256, 0, stream>>>(h_new, Wc, bc, out);
    }
}
